// SurrogateBlock_73753178407503
// MI455X (gfx1250) — matrix operand order repaired and checked by reading, not compiled, not run on hardware
//
#include <hip/hip_runtime.h>

// ---------------------------------------------------------------------------
// Types for CDNA5 WMMA (wave32): v16bf A/B operands, v8f accumulators.
// ---------------------------------------------------------------------------
typedef __bf16 bf16;
typedef __attribute__((ext_vector_type(8)))  bf16  bf16x8;
typedef __attribute__((ext_vector_type(16))) bf16  v16bf;
typedef __attribute__((ext_vector_type(8)))  float v8f;

typedef unsigned int u32x4 __attribute__((ext_vector_type(4)));
typedef int          i32x8 __attribute__((ext_vector_type(8)));
typedef int          i32x4 __attribute__((ext_vector_type(4)));

union FragB  { v16bf v; bf16x8 h[2]; };
union PackBF { bf16 h[2]; unsigned u; };

#define LN_EPS 1e-5f

#if __has_builtin(__builtin_amdgcn_tensor_load_to_lds) && __has_builtin(__builtin_amdgcn_s_wait_tensorcnt)
#define USE_TDM 1
#else
#define USE_TDM 0
#endif

__device__ __forceinline__ float fast_exp_clip(float x) {
    // matches reference: exp(clip(x, -30, 0))
    x = fmaxf(x, -30.0f);
    x = fminf(x, 0.0f);
    return __expf(x);
}

__device__ __forceinline__ float gelu_f(float x) {
    const float c = 0.7978845608028654f; // sqrt(2/pi)
    float t = tanhf(c * (x + 0.044715f * x * x * x));
    return 0.5f * x * (1.0f + t);
}

// xor-butterfly via ds_swizzle (group-of-32 mode: and=0x1f, or=0, xor=M)
template <int M>
__device__ __forceinline__ float swz_xor(float v) {
    return __int_as_float(
        __builtin_amdgcn_ds_swizzle(__float_as_int(v), (M << 10) | 0x1f));
}
__device__ __forceinline__ float half_red_max(float v) {
    v = fmaxf(v, swz_xor<1>(v));
    v = fmaxf(v, swz_xor<2>(v));
    v = fmaxf(v, swz_xor<4>(v));
    v = fmaxf(v, swz_xor<8>(v));
    return v;
}
__device__ __forceinline__ float half_red_sum(float v) {
    v += swz_xor<1>(v);
    v += swz_xor<2>(v);
    v += swz_xor<4>(v);
    v += swz_xor<8>(v);
    return v;
}

#if USE_TDM
// ---------------------------------------------------------------------------
// Tensor Data Mover: async 2D tile (bf16) global -> LDS, with optional LDS
// row padding. D# built per CDNA5 ISA 8.3-8.6: group0 {count=1, lds_addr,
// global_addr, type=2}; group1 {data_size=2B, pad ctl, dims/tile/stride};
// groups 2/3 + trailing 256b operand zero (2-D tensor). One issue per wave;
// EXEC-independent. 6-arg builtin form (clang-23 / therock-10.0 headers).
// ---------------------------------------------------------------------------
__device__ __forceinline__ void tdm_load_2d_bf16(
    unsigned lds_addr, unsigned long long gaddr,
    unsigned tile_d0, unsigned tile_d1,        // elements, rows
    unsigned long long d0_stride,              // row stride in elements
    unsigned pad_interval, unsigned pad_amount, unsigned pad_en)
{
    u32x4 g0;
    g0[0] = 1u;                                            // count=1, user mode
    g0[1] = lds_addr;                                      // LDS byte address
    g0[2] = (unsigned)(gaddr & 0xFFFFFFFFu);               // global_addr[31:0]
    g0[3] = (unsigned)((gaddr >> 32) & 0x1FFFFFFu) | 0x80000000u; // [56:32]|type=2

    i32x8 g1;
    unsigned w0 = (1u << 16);                              // data_size = 2 bytes
    if (pad_en) w0 |= (1u << 20) | (pad_interval << 22) | (pad_amount << 25);
    g1[0] = (int)w0;
    g1[1] = (int)((tile_d0 & 0xFFFFu) << 16);              // tensor_dim0 = tile_d0
    g1[2] = (int)(((tile_d0 >> 16) & 0xFFFFu) | ((tile_d1 & 0xFFFFu) << 16)); // tensor_dim1
    g1[3] = (int)(((tile_d1 >> 16) & 0xFFFFu) | ((tile_d0 & 0xFFFFu) << 16)); // tile_dim0
    g1[4] = (int)(tile_d1 & 0xFFFFu);                      // tile_dim1 (tile_dim2=0)
    g1[5] = (int)(unsigned)(d0_stride & 0xFFFFFFFFu);      // tensor_dim0_stride lo
    g1[6] = (int)(unsigned)((d0_stride >> 32) & 0xFFFFu);  // stride hi (dim1_stride=0)
    g1[7] = 0;

    i32x4 z4 = {0, 0, 0, 0};
    i32x8 z8 = {0, 0, 0, 0, 0, 0, 0, 0};
    __builtin_amdgcn_tensor_load_to_lds(g0, g1, z4, z4, z8, 0);
}
#endif

// ---------------------------------------------------------------------------
// LayerNorm (one block per row), f32 in -> bf16 out. rsqrtf ~ pwl_rsqrt.
// ---------------------------------------------------------------------------
__global__ __launch_bounds__(256)
void ln_bf16_kernel(const float* __restrict__ x, const float* __restrict__ w,
                    const float* __restrict__ b, bf16* __restrict__ out, int D)
{
    const int row = blockIdx.x;
    const float* xr = x + (size_t)row * D;
    bf16* orow = out + (size_t)row * D;
    __shared__ float red[256];
    const int tid = threadIdx.x;

    float s = 0.f;
    for (int i = tid; i < D; i += 256) s += xr[i];
    red[tid] = s; __syncthreads();
    for (int off = 128; off > 0; off >>= 1) { if (tid < off) red[tid] += red[tid + off]; __syncthreads(); }
    const float mu = red[0] / (float)D;
    __syncthreads();

    float v = 0.f;
    for (int i = tid; i < D; i += 256) { float d = xr[i] - mu; v += d * d; }
    red[tid] = v; __syncthreads();
    for (int off = 128; off > 0; off >>= 1) { if (tid < off) red[tid] += red[tid + off]; __syncthreads(); }
    const float inv = rsqrtf(red[0] / (float)D + LN_EPS);

    for (int i = tid; i < D; i += 256) {
        float y = (xr[i] - mu) * inv * w[i] + b[i];
        orow[i] = (bf16)y;
    }
}

// ---------------------------------------------------------------------------
// Tiled bf16 WMMA GEMM:  out[M,N] = A_bf16[M,K] @ W_f32[K,N] + bias (+epilogue)
// Block: 256 threads (8 waves), tile 128x128, K-step 32.
// A tile staged via TDM (one 16x32 slice per wave, padded LDS rows);
// W converted f32->bf16 in pk-pairs while staging transposed into LDS.
// ---------------------------------------------------------------------------
enum { EPI_BF16 = 0, EPI_BF16_GELU = 1, EPI_F32_RES = 2 };

template <int EPI>
__global__ __launch_bounds__(256)
void gemm_bf16_kernel(const bf16* __restrict__ A, const float* __restrict__ W,
                      const float* __restrict__ bias, const float* __restrict__ res,
                      void* __restrict__ outv, int M, int N, int K)
{
    constexpr int LDA = 48; // padded LDS row stride in bf16 elems (96B, 16B aligned)
    __shared__ __align__(16) bf16 Asub[128 * LDA];
    __shared__ __align__(16) bf16 Bsub[128 * LDA]; // transposed: [n][k]

    const int tid  = threadIdx.x;
    const int lane = tid & 31;
    const int wid  = tid >> 5;
    const int half = lane >> 4;
    const int l15  = lane & 15;

    const int n0 = blockIdx.x * 128;
    const int m0 = blockIdx.y * 128;
    const int wm = (wid >> 2) * 64; // wave M offset in tile
    const int wn = (wid & 3) * 32;  // wave N offset in tile

    v8f acc[4][2];
#pragma unroll
    for (int i = 0; i < 4; i++)
#pragma unroll
        for (int j = 0; j < 2; j++)
#pragma unroll
            for (int e = 0; e < 8; e++) acc[i][j][e] = 0.0f;

#if USE_TDM
    const unsigned asub_base = (unsigned)(unsigned long long)(uintptr_t)&Asub[0];
#else
    const int arow = tid >> 1;       // 0..127
    const int acol = (tid & 1) * 16; // 0,16
#endif
    const int bkp = tid >> 4;        // 0..15 : k-pair index
    const int bnb = (tid & 15) * 8;  // 0..120: n base
    unsigned* B32 = (unsigned*)Bsub;

    for (int k0 = 0; k0 < K; k0 += 32) {
#if USE_TDM
        // Async A tile: each wave DMAs rows [wid*16, wid*16+16) x 32 cols, with
        // LDS padding 16 DW rows + 8 DW pad -> stride 48 bf16.
        {
            const int wrow = wid * 16;
            tdm_load_2d_bf16(asub_base + (unsigned)(wrow * LDA * 2),
                             (unsigned long long)(uintptr_t)(A + (size_t)(m0 + wrow) * K + k0),
                             32u, 16u, (unsigned long long)K,
                             /*interval 16DW*/3u, /*pad 8DW*/7u, 1u);
        }
#else
        const bf16* ag = A + (size_t)(m0 + arow) * K + k0 + acol;
        *(bf16x8*)&Asub[arow * LDA + acol]     = *(const bf16x8*)ag;
        *(bf16x8*)&Asub[arow * LDA + acol + 8] = *(const bf16x8*)(ag + 8);
#endif
        // B tile 32x128 f32 -> bf16 pairs, transposed into Bsub[n][k]
        {
            const float* wg0 = W + (size_t)(k0 + 2 * bkp) * N + n0 + bnb;
            const float* wg1 = wg0 + N;
#pragma unroll
            for (int j = 0; j < 8; j++) {
                PackBF p;
                p.h[0] = (bf16)wg0[j];
                p.h[1] = (bf16)wg1[j];
                B32[(bnb + j) * (LDA / 2) + bkp] = p.u;
            }
        }
#if USE_TDM
        __builtin_amdgcn_s_wait_tensorcnt(0);
#endif
        __syncthreads();

#pragma unroll
        for (int mf = 0; mf < 4; mf++) {
            FragB a;
            const int ar = wm + mf * 16 + l15;
            const int kb = half * 8; // A layout: lanes0-15 K 0-7/16-23, lanes16-31 K 8-15/24-31
            a.h[0] = *(const bf16x8*)&Asub[ar * LDA + kb];
            a.h[1] = *(const bf16x8*)&Asub[ar * LDA + kb + 16];
#pragma unroll
            for (int nf = 0; nf < 2; nf++) {
                FragB bfr;
                const int bn  = wn + nf * 16 + l15;
                const int kb2 = half * 8; // B layout: lanes0-15 K 0-7/16-23, lanes16-31 K 8-15/24-31
                bfr.h[0] = *(const bf16x8*)&Bsub[bn * LDA + kb2];
                bfr.h[1] = *(const bf16x8*)&Bsub[bn * LDA + kb2 + 16];
                acc[mf][nf] = __builtin_amdgcn_wmma_f32_16x16x32_bf16(
                    false, a.v, false, bfr.v, (short)0, acc[mf][nf], false, false);
            }
        }
        __syncthreads();
    }

    // Epilogue. C layout: lanes0-15 -> (row=i, col=lane), lanes16-31 -> (row=i+8, col=lane-16)
#pragma unroll
    for (int mf = 0; mf < 4; mf++)
#pragma unroll
        for (int nf = 0; nf < 2; nf++) {
            const v8f c = acc[mf][nf];
            const int col = n0 + wn + nf * 16 + l15;
            const float bv = bias[col];
#pragma unroll
            for (int i = 0; i < 8; i++) {
                const int row = m0 + wm + mf * 16 + i + half * 8;
                float v = c[i] + bv;
                if constexpr (EPI == EPI_BF16_GELU) v = gelu_f(v);
                if constexpr (EPI == EPI_F32_RES) {
                    ((float*)outv)[(size_t)row * N + col] = v + res[(size_t)row * N + col];
                } else {
                    ((bf16*)outv)[(size_t)row * N + col] = (bf16)v;
                }
            }
        }
}

// ---------------------------------------------------------------------------
// Flash-style causal attention with surrogate softmax, all-WMMA.
// Grid: (T/64, H, B); block 128 (4 waves). Each wave owns 16 q rows,
// Q fragments register-resident (hd=128 -> 4 A-frags). K tile via TDM
// (8 rows per wave, padded LDS rows); V transposed into LDS in bf16 pairs.
// Online max/sum per row via ds_swizzle xor butterflies.
// ---------------------------------------------------------------------------
__global__ __launch_bounds__(128)
void attn_kernel(const bf16* __restrict__ qkv, // [B*T, 3*D] bf16
                 bf16* __restrict__ out)       // [B*T, D]   bf16
{
    constexpr int T = 1024, D = 2048, HD = 128, QKVS = 3 * D;
    constexpr int LDK = 144; // K tile row stride (288B, 16B aligned)
    constexpr int LDV = 48;  // Vt / P row stride (96B, 16B aligned)
    __shared__ __align__(16) bf16 Ksub[32 * LDK];   // [krow][feat]
    __shared__ __align__(16) bf16 Vt[HD * LDV];     // [feat][krow]
    __shared__ __align__(16) bf16 Psub[4][16 * LDV];

    const int tid = threadIdx.x, lane = tid & 31, wid = tid >> 5;
    const int half = lane >> 4, l15 = lane & 15;
    const int qt = blockIdx.x, hh = blockIdx.y, bb = blockIdx.z;
    const int rowbase = bb * T;
    const int q0 = qt * 64;
    const int qw = q0 + wid * 16;

    // Q fragments (A operand, 16x32 each)
    FragB qf[4];
    {
        const bf16* qp = qkv + (size_t)(rowbase + qw + l15) * QKVS + hh * HD;
#pragma unroll
        for (int kk = 0; kk < 4; kk++) {
            const int cb = kk * 32 + half * 8;
            qf[kk].h[0] = *(const bf16x8*)(qp + cb);
            qf[kk].h[1] = *(const bf16x8*)(qp + cb + 16);
        }
    }

    v8f O[8];
#pragma unroll
    for (int g = 0; g < 8; g++)
#pragma unroll
        for (int e = 0; e < 8; e++) O[g][e] = 0.0f;
    float mrow[8], lrow[8];
#pragma unroll
    for (int i = 0; i < 8; i++) { mrow[i] = -1e30f; lrow[i] = 0.0f; }

    const float scale = 0.088388347648318447f; // 1/sqrt(128)
    const int jmax = (q0 + 63) >> 5;

#if USE_TDM
    const unsigned ksub_base = (unsigned)(unsigned long long)(uintptr_t)&Ksub[0];
#endif
    const int vkp = tid >> 3;        // 0..15 : k-pair index
    const int vfb = (tid & 7) * 16;  // 0..112: feature base
    unsigned* V32 = (unsigned*)Vt;

    for (int j = 0; j <= jmax; j++) {
        // --- stage K (async TDM) and V-transposed (packed bf16 pairs) ---
#if USE_TDM
        {
            // each wave DMAs 8 K rows x 128 feats; pad 64 DW rows + 8 DW -> stride 144
            const int krow = wid * 8;
            tdm_load_2d_bf16(ksub_base + (unsigned)(krow * LDK * 2),
                             (unsigned long long)(uintptr_t)
                                 (qkv + (size_t)(rowbase + j * 32 + krow) * QKVS + D + hh * HD),
                             128u, 8u, (unsigned long long)QKVS,
                             /*interval 64DW*/5u, /*pad 8DW*/7u, 1u);
        }
#else
        {
            const int kr = tid >> 2;
            const int cb = (tid & 3) * 32;
            const bf16* kp = qkv + (size_t)(rowbase + j * 32 + kr) * QKVS + D + hh * HD + cb;
#pragma unroll
            for (int u = 0; u < 4; u++)
                *(bf16x8*)&Ksub[kr * LDK + cb + u * 8] = *(const bf16x8*)(kp + u * 8);
        }
#endif
        {
            const bf16* v0 = qkv + (size_t)(rowbase + j * 32 + 2 * vkp) * QKVS + 2 * D + hh * HD + vfb;
            const bf16* v1 = v0 + QKVS;
#pragma unroll
            for (int u = 0; u < 16; u++) {
                PackBF p;
                p.h[0] = v0[u];
                p.h[1] = v1[u];
                V32[(vfb + u) * (LDV / 2) + vkp] = p.u;
            }
        }
#if USE_TDM
        __builtin_amdgcn_s_wait_tensorcnt(0);
#endif
        __syncthreads();

        // --- S[16q x 32k] = Q @ K^T (two 16x16 n-tiles) ---
        v8f s[2];
#pragma unroll
        for (int t2 = 0; t2 < 2; t2++) {
#pragma unroll
            for (int e = 0; e < 8; e++) s[t2][e] = 0.0f;
#pragma unroll
            for (int kk = 0; kk < 4; kk++) {
                FragB kf;
                const int cb = kk * 32 + half * 8;
                kf.h[0] = *(const bf16x8*)&Ksub[(t2 * 16 + l15) * LDK + cb];
                kf.h[1] = *(const bf16x8*)&Ksub[(t2 * 16 + l15) * LDK + cb + 16];
                s[t2] = __builtin_amdgcn_wmma_f32_16x16x32_bf16(
                    false, qf[kk].v, false, kf.v, (short)0, s[t2], false, false);
            }
        }
        // scale + causal mask
#pragma unroll
        for (int t2 = 0; t2 < 2; t2++) {
            const int colg = j * 32 + t2 * 16 + l15;
#pragma unroll
            for (int i = 0; i < 8; i++) {
                const int rowg = qw + i + half * 8;
                const float v = s[t2][i] * scale;
                s[t2][i] = (colg > rowg) ? -1e30f : v;
            }
        }
        // --- online softmax (row reductions across each 16-lane half) ---
        float alpha[8], rs[8];
#pragma unroll
        for (int i = 0; i < 8; i++) {
            const float mx = half_red_max(fmaxf(s[0][i], s[1][i]));
            const float mn = fmaxf(mrow[i], mx);
            alpha[i] = fast_exp_clip(mrow[i] - mn);
            mrow[i] = mn;
            rs[i] = 0.0f;
        }
#pragma unroll
        for (int t2 = 0; t2 < 2; t2++) {
#pragma unroll
            for (int i = 0; i < 8; i++) {
                // masked entries get exp(-30), exactly like the reference clip
                const float p = fast_exp_clip(s[t2][i] - mrow[i]);
                rs[i] += p;
                Psub[wid][(i + half * 8) * LDV + t2 * 16 + l15] = (bf16)p;
            }
        }
#pragma unroll
        for (int i = 0; i < 8; i++)
            lrow[i] = lrow[i] * alpha[i] + half_red_sum(rs[i]);
#pragma unroll
        for (int g = 0; g < 8; g++)
#pragma unroll
            for (int i = 0; i < 8; i++) O[g][i] *= alpha[i];

        // --- O += P[16x32] @ V[32x128] (wave-local LDS round trip) ---
        FragB pf;
        pf.h[0] = *(const bf16x8*)&Psub[wid][l15 * LDV + half * 8];
        pf.h[1] = *(const bf16x8*)&Psub[wid][l15 * LDV + half * 8 + 16];
#pragma unroll
        for (int g = 0; g < 8; g++) {
            FragB vf;
            vf.h[0] = *(const bf16x8*)&Vt[(g * 16 + l15) * LDV + half * 8];
            vf.h[1] = *(const bf16x8*)&Vt[(g * 16 + l15) * LDV + half * 8 + 16];
            O[g] = __builtin_amdgcn_wmma_f32_16x16x32_bf16(
                false, pf.v, false, vf.v, (short)0, O[g], false, false);
        }
        __syncthreads();
    }

    // finalize: multiply by surrogate 1/l, store bf16
#pragma unroll
    for (int i = 0; i < 8; i++) lrow[i] = 1.0f / lrow[i];
#pragma unroll
    for (int g = 0; g < 8; g++)
#pragma unroll
        for (int i = 0; i < 8; i++) {
            const int row = rowbase + qw + i + half * 8;
            const int col = hh * HD + g * 16 + l15;
            out[(size_t)row * D + col] = (bf16)(O[g][i] * lrow[i]);
        }
}

// ---------------------------------------------------------------------------
// Orchestration
// ---------------------------------------------------------------------------
extern "C" void kernel_launch(void* const* d_in, const int* in_sizes, int n_in,
                              void* d_out, int out_size, void* d_ws, size_t ws_size,
                              hipStream_t stream)
{
    const float* x       = (const float*)d_in[0];
    const float* ln1w    = (const float*)d_in[1];
    const float* ln1b    = (const float*)d_in[2];
    const float* ln2w    = (const float*)d_in[3];
    const float* ln2b    = (const float*)d_in[4];
    const float* cattn_w = (const float*)d_in[5];
    const float* cattn_b = (const float*)d_in[6];
    const float* cproj_w = (const float*)d_in[7];
    const float* cproj_b = (const float*)d_in[8];
    const float* fc_w    = (const float*)d_in[9];
    const float* fc_b    = (const float*)d_in[10];
    const float* proj_w  = (const float*)d_in[11];
    const float* proj_b  = (const float*)d_in[12];

    const int BT = 2048, D = 2048, DF = 8192, QKV = 3 * D;

    char* ws   = (char*)d_ws;
    bf16*  ln1  = (bf16*)(ws);                        //  8 MB [BT, D]
    bf16*  qkv  = (bf16*)(ws + (size_t)( 8u << 20));  // 24 MB [BT, 3D]
    bf16*  attn = (bf16*)(ws + (size_t)(32u << 20));  //  8 MB [BT, D]
    float* x1   = (float*)(ws + (size_t)(40u << 20)); // 16 MB [BT, D]
    bf16*  ln2  = (bf16*)(ws + (size_t)(56u << 20));  //  8 MB [BT, D]
    bf16*  hbuf = (bf16*)(ws + (size_t)(64u << 20));  // 32 MB [BT, DF] (end 96 MB)

    // 1) ln1 = LN(x)
    ln_bf16_kernel<<<BT, 256, 0, stream>>>(x, ln1w, ln1b, ln1, D);
    // 2) qkv = ln1 @ c_attn_w + c_attn_b
    gemm_bf16_kernel<EPI_BF16><<<dim3(QKV / 128, BT / 128), 256, 0, stream>>>(
        ln1, cattn_w, cattn_b, nullptr, qkv, BT, QKV, D);
    // 3) attention (causal, surrogate softmax)
    attn_kernel<<<dim3(16, 16, 2), 128, 0, stream>>>(qkv, attn);
    // 4) x1 = attn @ c_proj_w + c_proj_b + x
    gemm_bf16_kernel<EPI_F32_RES><<<dim3(D / 128, BT / 128), 256, 0, stream>>>(
        attn, cproj_w, cproj_b, x, x1, BT, D, D);
    // 5) ln2 = LN(x1)
    ln_bf16_kernel<<<BT, 256, 0, stream>>>(x1, ln2w, ln2b, ln2, D);
    // 6) h = gelu(ln2 @ fc_w + fc_b)
    gemm_bf16_kernel<EPI_BF16_GELU><<<dim3(DF / 128, BT / 128), 256, 0, stream>>>(
        ln2, fc_w, fc_b, nullptr, hbuf, BT, DF, D);
    // 7) out = h @ proj_w + proj_b + x1
    gemm_bf16_kernel<EPI_F32_RES><<<dim3(D / 128, BT / 128), 256, 0, stream>>>(
        hbuf, proj_w, proj_b, x1, (float*)d_out, BT, D, DF);
}
